// EdgePooling_80642305950437
// MI455X (gfx1250) — matrix operand order repaired and checked by reading, not compiled, not run on hardware
//
#include <hip/hip_runtime.h>
#include <stdint.h>

#define IN_C 128
#define OUT_C 128
#define TM 64                 // rows per block in the WMMA GEMM
#define HASH_BITS 20
#define HASH_SIZE (1u << HASH_BITS)
#define HASH_MASK (HASH_SIZE - 1u)
#define EMPTY_KEY 0xFFFFFFFFu

typedef __bf16 bf16_t;
typedef __bf16 v16bf __attribute__((ext_vector_type(16)));
typedef __bf16 v8bf  __attribute__((ext_vector_type(8)));
typedef float  v8f   __attribute__((ext_vector_type(8)));

// ---------------------------------------------------------------- fills ----
__global__ void k_fill_f(float* p, long n, float v) {
    long i = (long)blockIdx.x * blockDim.x + threadIdx.x;
    if (i < n) p[i] = v;
}
__global__ void k_fill_u32(uint32_t* p, long n, uint32_t v) {
    long i = (long)blockIdx.x * blockDim.x + threadIdx.x;
    if (i < n) p[i] = v;
}

// ------------------------------------------- u = W_t @ W_s, c0 = b_t.W_s + b_s
__global__ void k_wvec(const float* __restrict__ Wt, const float* __restrict__ bt,
                       const float* __restrict__ Ws, const float* __restrict__ bs,
                       float* __restrict__ u, float* __restrict__ c0) {
    int i = threadIdx.x;  // 128 threads
    if (i < IN_C) {
        float acc = 0.f;
        for (int j = 0; j < OUT_C; ++j) acc += Wt[i * OUT_C + j] * Ws[j];
        u[i] = acc;
    }
    if (i == 0) {
        float c = bs[0];
        for (int j = 0; j < OUT_C; ++j) c += bt[j] * Ws[j];
        c0[0] = c;
    }
}

// ------------------------------- e[i] = sigmoid((x[src]+x[dst]) . u + c0) ----
__global__ __launch_bounds__(256) void k_scores(const float* __restrict__ x,
                                                const int* __restrict__ src,
                                                const int* __restrict__ dst,
                                                const float* __restrict__ u,
                                                const float* __restrict__ c0,
                                                float* __restrict__ e_out, int E) {
    __shared__ float us[IN_C];
    if (threadIdx.x < IN_C) us[threadIdx.x] = u[threadIdx.x];
    __syncthreads();
    int i = blockIdx.x * 256 + threadIdx.x;
    if (i >= E) return;
    int s = src[i], d = dst[i];
    const float4* xs = (const float4*)(x + (size_t)s * IN_C);
    const float4* xd = (const float4*)(x + (size_t)d * IN_C);
    const float4* uv = (const float4*)us;
    float acc = 0.f;
    for (int q = 0; q < IN_C / 4; ++q) {
        float4 a = xs[q], b = xd[q], w = uv[q];
        acc += (a.x + b.x) * w.x + (a.y + b.y) * w.y +
               (a.z + b.z) * w.z + (a.w + b.w) * w.w;
    }
    float z = acc + c0[0];
    e_out[i] = 1.0f / (1.0f + __expf(-z));
}

// ---- pack sortable keys: descending score, ascending index on ties ---------
__global__ void k_keys(const float* __restrict__ e, uint64_t* __restrict__ keys,
                       int E, int Epad) {
    int i = blockIdx.x * blockDim.x + threadIdx.x;
    if (i >= Epad) return;
    if (i >= E) { keys[i] = ~0ull; return; }
    uint32_t b = __float_as_uint(e[i]);
    uint32_t s = b ^ ((b >> 31) ? 0xFFFFFFFFu : 0x80000000u);  // ascending-sortable
    keys[i] = ((uint64_t)(~s) << 32) | (uint32_t)i;            // sort ascending
}

__global__ void k_bitonic(uint64_t* __restrict__ a, int j, int k) {
    int i = blockIdx.x * blockDim.x + threadIdx.x;
    int ixj = i ^ j;
    if (ixj > i) {
        uint64_t xi = a[i], xj = a[ixj];
        bool up = ((i & k) == 0);
        if ((xi > xj) == up) { a[i] = xj; a[ixj] = xi; }
    }
}

// --------------- sequential greedy matching over sorted edges (serial) ------
__global__ void k_greedy(const uint64_t* __restrict__ sorted,
                         const int* __restrict__ src, const int* __restrict__ dst,
                         const float* __restrict__ e, const float* __restrict__ rnd,
                         int* __restrict__ cluster_i, int* __restrict__ rowA,
                         int* __restrict__ rowB, float* __restrict__ out_score,
                         float* __restrict__ out_nc, int* __restrict__ nclus_ws,
                         int N, int E) {
    __shared__ uint32_t avail[1024];  // 32768 node bits
    if (threadIdx.x != 0) return;
    int nw = (N + 31) / 32;
    for (int i = 0; i < nw; ++i) avail[i] = 0xFFFFFFFFu;
    int next = 0;
    for (int i = 0; i < E; ++i) {
        uint32_t idx = (uint32_t)(sorted[i] & 0xFFFFFFFFull);
        float sc = e[idx];
        bool acc = rnd[idx] <= sc;
        int s = src[idx], d = dst[idx];
        uint32_t ms = (avail[s >> 5] >> (s & 31)) & 1u;
        uint32_t md = (avail[d >> 5] >> (d & 31)) & 1u;
        if (acc && ms && md) {
            avail[s >> 5] &= ~(1u << (s & 31));
            avail[d >> 5] &= ~(1u << (d & 31));
            int c = next++;
            cluster_i[s] = c; cluster_i[d] = c;
            rowA[c] = s; rowB[c] = d;
            out_score[c] = sc;
        }
    }
    int r = 0;
    for (int v = 0; v < N; ++v) {
        if (cluster_i[v] < 0) {
            int c = next + r; ++r;
            cluster_i[v] = c; rowA[c] = v; rowB[c] = v;
        }
    }
    int nc = next + r;
    *nclus_ws = nc;
    *out_nc = (float)nc;
}

// -------- per-node outputs: cluster (float), new_batch scatter --------------
__global__ void k_map(const int* __restrict__ cluster_i, const int* __restrict__ rowA,
                      const int* __restrict__ rowB, const int* __restrict__ batch,
                      float* __restrict__ out_cluster, float* __restrict__ out_batch,
                      int N) {
    int v = blockIdx.x * blockDim.x + threadIdx.x;
    if (v >= N) return;
    int c = cluster_i[v];
    out_cluster[v] = (float)c;
    int a = rowA[c], b = rowB[c];
    int winner = a > b ? a : b;           // deterministic writer per cluster
    if (v == winner) out_batch[c] = (float)batch[v];
}

// -------- WMMA pooled GEMM: new_x[c] = (x[ra]+x[rb]) @ W_t + b_t ------------
// Block: 256 threads = 8 waves; TM=64 rows x 128 cols per block.
// W_t staged TRANSPOSED in LDS (Bt[col][k], padded rows) so each lane's B
// fragment is one contiguous 32B run -> ds_load_b128 pairs, no scalar packing.
__global__ __launch_bounds__(256) void k_pool_gemm(const float* __restrict__ x,
                                                   const float* __restrict__ Wt,
                                                   const float* __restrict__ bt,
                                                   const int* __restrict__ rowA,
                                                   const int* __restrict__ rowB,
                                                   const int* __restrict__ nclus_ws,
                                                   float* __restrict__ newx) {
    __shared__ bf16_t Bt[OUT_C][IN_C + 8];  // transposed W_t [col][k], padded
    __shared__ bf16_t As[TM][IN_C];         // A tile bf16
    __shared__ int ra[TM], rb[TM];
    int nc = *nclus_ws;
    int base = blockIdx.x * TM;
    if (base >= nc) return;  // block-uniform: EXEC stays all-ones for WMMA
    int t = threadIdx.x;
    // transpose-load W_t -> bf16 LDS (padded rows avoid write bank conflicts)
    for (int q = t; q < IN_C * OUT_C; q += 256) {
        int k = q >> 7, c = q & 127;
        Bt[c][k] = (bf16_t)Wt[q];
    }
    if (t < TM) {
        int r = base + t;
        int a = 0, b = 0;
        if (r < nc) { a = rowA[r]; b = rowB[r]; }
        ra[t] = a; rb[t] = b;
    }
    __syncthreads();
    for (int q = t; q < TM * IN_C; q += 256) {
        int m = q >> 7, c = q & 127;
        As[m][c] = (bf16_t)(x[(size_t)ra[m] * IN_C + c] + x[(size_t)rb[m] * IN_C + c]);
    }
    __syncthreads();

    int wave = t >> 5, lane = t & 31;
    int half = lane >> 4;                 // half-wave selector
    int col  = (wave << 4) + (lane & 15); // this wave's 16-col strip
    int mlo  = lane & 15;                 // A row within 16-row tile

    float bias = bt[col];
    v8f acc[TM / 16];
#pragma unroll
    for (int mt = 0; mt < TM / 16; ++mt)
#pragma unroll
        for (int r = 0; r < 8; ++r) acc[mt][r] = bias;  // C = broadcast bias

#pragma unroll
    for (int kk = 0; kk < IN_C; kk += 32) {
        // B 32x16 bf16: lanes 0-15 K=kk..kk+15, lanes 16-31 K=kk+16..kk+31;
        // contiguous in transposed layout -> single 32B vector load
        v16bf b = *(const v16bf*)(&Bt[col][kk + 16 * half]);
#pragma unroll
        for (int mt = 0; mt < TM / 16; ++mt) {
            const bf16_t* arow = &As[mt * 16 + mlo][0];
            // A 16x32 bf16: per lane two contiguous 8-element runs
            v8bf alo = *(const v8bf*)(arow + kk + half * 16);       // K=kk+16h..+7
            v8bf ahi = *(const v8bf*)(arow + kk + 8 + half * 16);   // K=kk+16h+8..+15
            v16bf a = __builtin_shufflevector(alo, ahi, 0, 1, 2, 3, 4, 5, 6, 7,
                                              8, 9, 10, 11, 12, 13, 14, 15);
            acc[mt] = __builtin_amdgcn_wmma_f32_16x16x32_bf16(
                false, a, false, b, (short)0, acc[mt], false, false);
        }
    }
#pragma unroll
    for (int mt = 0; mt < TM / 16; ++mt) {
#pragma unroll
        for (int r = 0; r < 8; ++r) {
            int grow = base + mt * 16 + r + half * 8;  // D: VGPR r -> M=r / r+8
            if (grow < nc) newx[(size_t)grow * OUT_C + col] = acc[mt][r];
        }
    }
}

// -------- new_ei + hash insert (min original index per coalesce key) --------
__global__ void k_newei(const int* __restrict__ src, const int* __restrict__ dst,
                        const int* __restrict__ cluster_i, float* __restrict__ out_ei,
                        uint32_t* __restrict__ hkey, uint32_t* __restrict__ hmin,
                        int E, int N) {
    int i = blockIdx.x * blockDim.x + threadIdx.x;
    if (i >= E) return;
    int ns = cluster_i[src[i]], nd = cluster_i[dst[i]];
    out_ei[i] = (float)ns;
    out_ei[(size_t)E + i] = (float)nd;
    uint32_t key = (uint32_t)ns * (uint32_t)N + (uint32_t)nd;  // < 2^30
    uint32_t h = (key * 2654435761u) >> (32 - HASH_BITS);
    for (;;) {
        uint32_t prev = atomicCAS(&hkey[h], EMPTY_KEY, key);
        if (prev == EMPTY_KEY || prev == key) break;
        h = (h + 1) & HASH_MASK;
    }
    atomicMin((unsigned int*)&hmin[h], (unsigned int)i);
}

__global__ void k_valid(const int* __restrict__ src, const int* __restrict__ dst,
                        const int* __restrict__ cluster_i,
                        const uint32_t* __restrict__ hkey,
                        const uint32_t* __restrict__ hmin,
                        float* __restrict__ out_valid, int E, int N) {
    int i = blockIdx.x * blockDim.x + threadIdx.x;
    if (i >= E) return;
    int ns = cluster_i[src[i]], nd = cluster_i[dst[i]];
    uint32_t key = (uint32_t)ns * (uint32_t)N + (uint32_t)nd;
    uint32_t h = (key * 2654435761u) >> (32 - HASH_BITS);
    while (hkey[h] != key) h = (h + 1) & HASH_MASK;
    out_valid[i] = ((ns != nd) && (hmin[h] == (uint32_t)i)) ? 1.0f : 0.0f;
}

// ---------------------------------------------------------------------------
extern "C" void kernel_launch(void* const* d_in, const int* in_sizes, int n_in,
                              void* d_out, int out_size, void* d_ws, size_t ws_size,
                              hipStream_t stream) {
    const float* x   = (const float*)d_in[0];
    const int*   ei  = (const int*)d_in[1];
    const int*   bat = (const int*)d_in[2];
    const float* rnd = (const float*)d_in[3];
    const float* Wt  = (const float*)d_in[4];
    const float* bt  = (const float*)d_in[5];
    const float* Ws  = (const float*)d_in[6];
    const float* bs  = (const float*)d_in[7];

    const int N = in_sizes[0] / IN_C;   // 32768
    const int E = in_sizes[1] / 2;      // 262144
    const int* src = ei;
    const int* dst = ei + E;
    float* out = (float*)d_out;

    // output regions (flat f32 concat, reference return order)
    size_t off_newx    = 0;
    size_t off_cluster = (size_t)N * OUT_C;
    size_t off_batch   = off_cluster + N;
    size_t off_score   = off_batch + N;
    size_t off_nc      = off_score + N;
    size_t off_ei      = off_nc + 1;
    size_t off_valid   = off_ei + 2 * (size_t)E;
    size_t off_e       = off_valid + E;

    // workspace carving
    char* w = (char*)d_ws;
    float*    u         = (float*)w;                 // 128
    float*    c0        = u + IN_C;                  // 1
    int*      nclus     = (int*)(w + 1024);
    int*      cluster_i = (int*)(w + 4096);
    int*      rowA      = cluster_i + N;
    int*      rowB      = rowA + N;
    uint32_t* hkey      = (uint32_t*)(rowB + N);
    uint32_t* hmin      = hkey + HASH_SIZE;
    int Epad = 1; while (Epad < E) Epad <<= 1;
    uint64_t* keys = (uint64_t*)(((uintptr_t)(hmin + HASH_SIZE) + 255) & ~(uintptr_t)255);

    const int B = 256;
    // init (d_out is poisoned; every region must be written)
    k_fill_f<<<((long)N * OUT_C + B - 1) / B, B, 0, stream>>>(out + off_newx, (long)N * OUT_C, 0.f);
    k_fill_f<<<(N + B - 1) / B, B, 0, stream>>>(out + off_batch, N, 0.f);
    k_fill_f<<<(N + B - 1) / B, B, 0, stream>>>(out + off_score, N, 1.f);
    k_fill_u32<<<(HASH_SIZE + B - 1) / B, B, 0, stream>>>(hkey, HASH_SIZE, EMPTY_KEY);
    k_fill_u32<<<(HASH_SIZE + B - 1) / B, B, 0, stream>>>(hmin, HASH_SIZE, 0xFFFFFFFFu);
    k_fill_u32<<<(N + B - 1) / B, B, 0, stream>>>((uint32_t*)cluster_i, N, 0xFFFFFFFFu);

    // scores via collapsed GEMV (u = W_t @ W_s)
    k_wvec<<<1, 128, 0, stream>>>(Wt, bt, Ws, bs, u, c0);
    k_scores<<<(E + B - 1) / B, B, 0, stream>>>(x, src, dst, u, c0, out + off_e, E);

    // stable descending argsort via bitonic on packed keys
    k_keys<<<(Epad + B - 1) / B, B, 0, stream>>>(out + off_e, keys, E, Epad);
    for (int k = 2; k <= Epad; k <<= 1)
        for (int j = k >> 1; j > 0; j >>= 1)
            k_bitonic<<<(Epad + B - 1) / B, B, 0, stream>>>(keys, j, k);

    // sequential greedy matching + remaining-node cluster ids
    k_greedy<<<1, 32, 0, stream>>>(keys, src, dst, out + off_e, rnd, cluster_i,
                                   rowA, rowB, out + off_score, out + off_nc,
                                   nclus, N, E);

    // per-node outputs
    k_map<<<(N + B - 1) / B, B, 0, stream>>>(cluster_i, rowA, rowB, bat,
                                             out + off_cluster, out + off_batch, N);

    // pooled features: the WMMA GEMM (only num_clusters rows, never full E GEMM)
    k_pool_gemm<<<(N + TM - 1) / TM, 256, 0, stream>>>(x, Wt, bt, rowA, rowB, nclus,
                                                       out + off_newx);

    // remapped edges + coalesce validity
    k_newei<<<(E + B - 1) / B, B, 0, stream>>>(src, dst, cluster_i, out + off_ei,
                                               hkey, hmin, E, N);
    k_valid<<<(E + B - 1) / B, B, 0, stream>>>(src, dst, cluster_i, hkey, hmin,
                                               out + off_valid, E, N);
}
